// HybridAttention_72112500900083
// MI455X (gfx1250) — matrix operand order repaired and checked by reading, not compiled, not run on hardware
//
#include <hip/hip_runtime.h>
#include <hip/hip_bf16.h>

typedef __bf16 bf16_t;
typedef __attribute__((ext_vector_type(16))) __bf16 v16bf;
typedef __attribute__((ext_vector_type(8)))  __bf16 v8bf;
typedef __attribute__((ext_vector_type(8)))  float  v8f;
typedef __attribute__((ext_vector_type(4)))  int    v4i;

#define DEVINL __device__ __forceinline__

#if defined(__gfx1250__) && __has_builtin(__builtin_amdgcn_global_load_async_to_lds_b128)
#define ATHENA_ASYNC_LDS 1
typedef __attribute__((address_space(1))) v4i* gvec_t;   // global source (b128)
typedef __attribute__((address_space(3))) v4i* lvec_t;   // LDS destination
#endif

DEVINL float  bf2f(bf16_t b) { return (float)b; }
DEVINL bf16_t f2bf(float f)  { return (bf16_t)f; }
DEVINL v8f    zero8()        { v8f z = {0.f,0.f,0.f,0.f,0.f,0.f,0.f,0.f}; return z; }

DEVINL void wait_async0() {
#if defined(ATHENA_ASYNC_LDS)
#if __has_builtin(__builtin_amdgcn_s_wait_asynccnt)
  __builtin_amdgcn_s_wait_asynccnt(0);
#else
  asm volatile("s_wait_asynccnt 0x0" ::: "memory");
#endif
#endif
}

// ---------------------------------------------------------------------------
// WMMA operand loaders (gfx1250 wave32; A/B use this file's own k order, same for both; C/D per ISA):
// A (16x32 bf16): lane L (half=L>>4, r=L&15) holds row (row0+r):
//   elems 0..15 = K[k0+16*half .. +15]  (contiguous)
// B (32x16 bf16): lane L holds column n=(L&15) (i.e. source row row0+(L&15)),
//   elems 0..15 = K[k0+16*(L>>4) .. +15]  (contiguous)
// C/D (16x16 f32): lane L, vgpr r -> (M = r + 8*(L>>4), N = L&15)
// ---------------------------------------------------------------------------
DEVINL v16bf load_a(const bf16_t* base, int ld, int row0, int k0, int lane) {
  int half = (lane >> 4) & 1, r = lane & 15;
  const bf16_t* p = base + (size_t)(row0 + r) * ld + k0 + 16 * half;
  v8bf lo = *(const v8bf*)p;
  v8bf hi = *(const v8bf*)(p + 8);
  return __builtin_shufflevector(lo, hi, 0,1,2,3,4,5,6,7,8,9,10,11,12,13,14,15);
}

DEVINL v16bf load_b(const bf16_t* base, int ld, int row0, int k0, int lane) {
  const bf16_t* p = base + (size_t)(row0 + (lane & 15)) * ld + k0 + 16 * ((lane >> 4) & 1);
  return *(const v16bf*)p;
}

DEVINL v8f wmma_bf16(v16bf a, v16bf b, v8f c) {
  return __builtin_amdgcn_wmma_f32_16x16x32_bf16(false, a, false, b, (short)0, c,
                                                 false, false);
}

// ---------------------------------------------------------------------------
// f32 -> bf16 conversion (weights)
// ---------------------------------------------------------------------------
__global__ void cvt_bf16(const float* __restrict__ in, bf16_t* __restrict__ out, int n) {
  int i = blockIdx.x * blockDim.x + threadIdx.x;
  if (i < n) out[i] = f2bf(in[i]);
}

// ---------------------------------------------------------------------------
// Fused dual LayerNorm: writes bf16 xg (global path, full 8192x1024) and
// bf16 xl_loc (local path, only rows (b, s<8) packed as [b*8+s][1024]).
// ---------------------------------------------------------------------------
__global__ __launch_bounds__(256) void ln_kernel(
    const float* __restrict__ x,
    const float* __restrict__ lg, const float* __restrict__ lb,
    const float* __restrict__ gg, const float* __restrict__ gb,
    bf16_t* __restrict__ xg, bf16_t* __restrict__ xloc)
{
  __shared__ float s1[256], s2[256];
  int row = blockIdx.x, tid = threadIdx.x;
  const float* xr = x + (size_t)row * 1024;
  float a = 0.f, b2 = 0.f;
#pragma unroll
  for (int k = 0; k < 4; ++k) { float v = xr[tid + 256*k]; a += v; b2 += v*v; }
  s1[tid] = a; s2[tid] = b2;
  __syncthreads();
  for (int s = 128; s > 0; s >>= 1) {
    if (tid < s) { s1[tid] += s1[tid + s]; s2[tid] += s2[tid + s]; }
    __syncthreads();
  }
  float mean = s1[0] * (1.f/1024.f);
  float var  = s2[0] * (1.f/1024.f) - mean*mean;
  float rstd = rsqrtf(var + 1e-5f);
  int bb = row >> 10, ss = row & 1023;
#pragma unroll
  for (int k = 0; k < 4; ++k) {
    int c = tid + 256*k;
    float v = (xr[c] - mean) * rstd;
    xg[(size_t)row * 1024 + c] = f2bf(v * gg[c] + gb[c]);
    if (ss < 8) xloc[(size_t)(bb*8 + ss) * 1024 + c] = f2bf(v * lg[c] + lb[c]);
  }
}

// ---------------------------------------------------------------------------
// QKV GEMM: Y[M,3072] = X[M,1024] @ W[3072,1024]^T + bias, split into q/k/v.
// q columns scaled by dh^-0.5. If vt!=0, V written transposed [b][h][d][s].
// Wave tile: 32x64 (2x4 C tiles), 8 wmma per 32-wide K step.
// ---------------------------------------------------------------------------
__global__ __launch_bounds__(256) void gemm_qkv(
    const bf16_t* __restrict__ X, const bf16_t* __restrict__ W,
    const float* __restrict__ bias,
    bf16_t* __restrict__ qout, bf16_t* __restrict__ kout, bf16_t* __restrict__ vout,
    int M, int N, int K, float qscale, int vt)
{
  int lane = threadIdx.x & 31;
  int wave = (blockIdx.x * blockDim.x + threadIdx.x) >> 5;
  int ntn = N >> 6;
  int mt = wave / ntn, nt = wave % ntn;
  int m0 = mt * 32, n0 = nt * 64;
  if (m0 >= M) return;

  v8f acc[2][4];
#pragma unroll
  for (int i = 0; i < 2; ++i)
#pragma unroll
    for (int j = 0; j < 4; ++j) acc[i][j] = zero8();

  const bf16_t* arow0 = X + (size_t)(m0 + (lane & 15)) * K;
  const bf16_t* arow1 = X + (size_t)(m0 + 16 + (lane & 15)) * K;
  for (int k0 = 0; k0 < K; k0 += 32) {
    if (k0 + 128 < K) {                       // pull next K-blocks into cache
      __builtin_prefetch(arow0 + k0 + 128, 0, 1);
      __builtin_prefetch(arow1 + k0 + 128, 0, 1);
    }
    v16bf a0 = load_a(X, K, m0,      k0, lane);
    v16bf a1 = load_a(X, K, m0 + 16, k0, lane);
#pragma unroll
    for (int j = 0; j < 4; ++j) {
      v16bf b = load_b(W, K, n0 + 16*j, k0, lane);
      acc[0][j] = wmma_bf16(a0, b, acc[0][j]);
      acc[1][j] = wmma_bf16(a1, b, acc[1][j]);
    }
  }

  int half = lane >> 4, col = lane & 15;
#pragma unroll
  for (int j = 0; j < 4; ++j) {
    int nbase = n0 + 16*j;                    // wave-uniform
    int seg = nbase >> 10;                    // 0=q, 1=k, 2=v (64 | 1024)
    int n = nbase + col;
    float bn = bias[n];
    if (seg == 0) {
#pragma unroll
      for (int i = 0; i < 2; ++i)
#pragma unroll
        for (int r = 0; r < 8; ++r) {
          int m = m0 + 16*i + r + 8*half;
          qout[(size_t)m * 1024 + n] = f2bf((acc[i][j][r] + bn) * qscale);
        }
    } else if (seg == 1) {
#pragma unroll
      for (int i = 0; i < 2; ++i)
#pragma unroll
        for (int r = 0; r < 8; ++r) {
          int m = m0 + 16*i + r + 8*half;
          kout[(size_t)m * 1024 + (n - 1024)] = f2bf(acc[i][j][r] + bn);
        }
    } else if (vt) {
      int hh = (n - 2048) >> 6, d = (n - 2048) & 63;
#pragma unroll
      for (int i = 0; i < 2; ++i)
#pragma unroll
        for (int r = 0; r < 8; ++r) {
          int m = m0 + 16*i + r + 8*half;
          int bb = m >> 10, ss = m & 1023;
          vout[(size_t)((bb*16 + hh)*64 + d) * 1024 + ss] = f2bf(acc[i][j][r] + bn);
        }
    } else {
#pragma unroll
      for (int i = 0; i < 2; ++i)
#pragma unroll
        for (int r = 0; r < 8; ++r) {
          int m = m0 + 16*i + r + 8*half;
          vout[(size_t)m * 1024 + (n - 2048)] = f2bf(acc[i][j][r] + bn);
        }
    }
  }
}

// ---------------------------------------------------------------------------
// Out-projection GEMM: out[M,1024] (f32) = X @ W^T + bias.
// scatter_add: map packed local row m=(b*8+s) -> d_out row b*1024+s, accumulate.
// ---------------------------------------------------------------------------
__global__ __launch_bounds__(256) void gemm_proj(
    const bf16_t* __restrict__ X, const bf16_t* __restrict__ W,
    const float* __restrict__ bias, float* __restrict__ out,
    int M, int N, int K, int scatter_add)
{
  int lane = threadIdx.x & 31;
  int wave = (blockIdx.x * blockDim.x + threadIdx.x) >> 5;
  int ntn = N >> 6;
  int mt = wave / ntn, nt = wave % ntn;
  int m0 = mt * 32, n0 = nt * 64;
  if (m0 >= M) return;

  v8f acc[2][4];
#pragma unroll
  for (int i = 0; i < 2; ++i)
#pragma unroll
    for (int j = 0; j < 4; ++j) acc[i][j] = zero8();

  const bf16_t* arow0 = X + (size_t)(m0 + (lane & 15)) * K;
  const bf16_t* arow1 = X + (size_t)(m0 + 16 + (lane & 15)) * K;
  for (int k0 = 0; k0 < K; k0 += 32) {
    if (k0 + 128 < K) {
      __builtin_prefetch(arow0 + k0 + 128, 0, 1);
      __builtin_prefetch(arow1 + k0 + 128, 0, 1);
    }
    v16bf a0 = load_a(X, K, m0,      k0, lane);
    v16bf a1 = load_a(X, K, m0 + 16, k0, lane);
#pragma unroll
    for (int j = 0; j < 4; ++j) {
      v16bf b = load_b(W, K, n0 + 16*j, k0, lane);
      acc[0][j] = wmma_bf16(a0, b, acc[0][j]);
      acc[1][j] = wmma_bf16(a1, b, acc[1][j]);
    }
  }

  int half = lane >> 4, col = lane & 15;
#pragma unroll
  for (int j = 0; j < 4; ++j) {
    int n = n0 + 16*j + col;
    float bn = bias[n];
#pragma unroll
    for (int i = 0; i < 2; ++i) {
#pragma unroll
      for (int r = 0; r < 8; ++r) {
        int m = m0 + 16*i + r + 8*half;
        float val = acc[i][j][r] + bn;
        size_t row = scatter_add ? (size_t)((m >> 3) * 1024 + (m & 7)) : (size_t)m;
        float* p = out + row * (size_t)N + n;
        if (scatter_add) *p += val; else *p = val;
      }
    }
  }
}

// ---------------------------------------------------------------------------
// Global attention (flash-style), one block per (b, h, 128-query slab),
// 8 waves x 16 queries. Computes S^T = K_blk x Q^T so each query's scores
// live in-lane (lane = query): online softmax needs only a shfl_xor(16)
// pair-merge, and P^T feeds the O^T = V^T x P^T WMMA as the B operand.
// K/V blocks (32 keys) staged in LDS via GLOBAL_LOAD_ASYNC_TO_LDS (ASYNCcnt)
// when the toolchain exposes the builtin, else plain load+ds_store.
// ---------------------------------------------------------------------------
__global__ __launch_bounds__(256) void attn_global(
    const bf16_t* __restrict__ Q, const bf16_t* __restrict__ Kc,
    const bf16_t* __restrict__ VT, bf16_t* __restrict__ O)
{
  __shared__ bf16_t Kblk[32 * 64];   // [key][d]
  __shared__ bf16_t Vblk[64 * 32];   // [d][key]  (from transposed V buffer)

  int tid = threadIdx.x, lane = tid & 31, wave = tid >> 5;
  int qblk = blockIdx.x & 7;
  int bh   = blockIdx.x >> 3;          // 0..127
  int b = bh >> 4, h = bh & 15;
  int q0 = qblk * 128 + wave * 16;

  const bf16_t* qbase = Q + ((size_t)(b * 1024 + q0)) * 1024 + h * 64;
  v16bf bq0 = load_b(qbase, 1024, 0, 0,  lane);   // B = Q^T, d = 0..31
  v16bf bq1 = load_b(qbase, 1024, 0, 32, lane);   // B = Q^T, d = 32..63

  v8f ot[4];
#pragma unroll
  for (int t = 0; t < 4; ++t) ot[t] = zero8();
  float m_run = -__builtin_inff(), l_run = 0.f;

  const bf16_t* krow = Kc + ((size_t)b * 1024) * 1024 + h * 64;
  const bf16_t* vrow = VT + ((size_t)((b * 16 + h) * 64)) * 1024;

  int kk = (tid * 8) >> 6, ck = (tid * 8) & 63;       // K-staging coords
  int dv = tid >> 2,       cv = (tid & 3) * 8;        // V-staging coords

  for (int kb = 0; kb < 1024; kb += 32) {
    __syncthreads();                              // protect previous LDS reads
#if defined(ATHENA_ASYNC_LDS)
    __builtin_amdgcn_global_load_async_to_lds_b128(
        (gvec_t)(krow + (size_t)(kb + kk) * 1024 + ck),
        (lvec_t)(Kblk + tid * 8), 0, 0);
    __builtin_amdgcn_global_load_async_to_lds_b128(
        (gvec_t)(vrow + (size_t)dv * 1024 + kb + cv),
        (lvec_t)(Vblk + dv * 32 + cv), 0, 0);
    wait_async0();
#else
    *(v8bf*)(Kblk + tid * 8)      = *(const v8bf*)(krow + (size_t)(kb + kk) * 1024 + ck);
    *(v8bf*)(Vblk + dv * 32 + cv) = *(const v8bf*)(vrow + (size_t)dv * 1024 + kb + cv);
#endif
    __syncthreads();

    // S^T tiles: tile0 = keys kb..kb+15, tile1 = keys kb+16..kb+31.
    v8f st0 = zero8(), st1 = zero8();
    st0 = wmma_bf16(load_a(Kblk, 64, 0,  0,  lane), bq0, st0);
    st0 = wmma_bf16(load_a(Kblk, 64, 0,  32, lane), bq1, st0);
    st1 = wmma_bf16(load_a(Kblk, 64, 16, 0,  lane), bq0, st1);
    st1 = wmma_bf16(load_a(Kblk, 64, 16, 32, lane), bq1, st1);

    // Online softmax. Lane L and L^16 together cover all 32 keys of q=(L&15).
    float mx = st0[0];
#pragma unroll
    for (int r = 0; r < 8; ++r) { mx = fmaxf(mx, st0[r]); mx = fmaxf(mx, st1[r]); }
    mx = fmaxf(mx, __shfl_xor(mx, 16));
    float m_new = fmaxf(m_run, mx);
    float corr  = __expf(m_run - m_new);

    v8bf pt0, pt1;
    float ls = 0.f;
#pragma unroll
    for (int r = 0; r < 8; ++r) {
      float p0 = __expf(st0[r] - m_new);
      float p1 = __expf(st1[r] - m_new);
      pt0[r] = f2bf(p0); pt1[r] = f2bf(p1);
      ls += p0 + p1;
    }
    ls += __shfl_xor(ls, 16);
    l_run = l_run * corr + ls;
    m_run = m_new;

#pragma unroll
    for (int t = 0; t < 4; ++t)
#pragma unroll
      for (int r = 0; r < 8; ++r) ot[t][r] = ot[t][r] * corr;

    // Build B operand P^T (32 keys x 16 q): swap half-rows with lane^16.
    union { v8bf v; unsigned u[4]; } x0, x1;
    x0.v = pt0; x1.v = pt1;
#pragma unroll
    for (int i = 0; i < 4; ++i) {
      x0.u[i] = __shfl_xor(x0.u[i], 16);
      x1.u[i] = __shfl_xor(x1.u[i], 16);
    }
    bool up = (lane >> 4) & 1;
    v8bf lo = up ? x1.v : pt0;
    v8bf hi = up ? pt1  : x0.v;
    v16bf pb = __builtin_shufflevector(lo, hi, 0,1,2,3,4,5,6,7,8,9,10,11,12,13,14,15);

    // O^T += V^T x P^T  (A = V^T tiles: rows d, K = keys)
#pragma unroll
    for (int t = 0; t < 4; ++t)
      ot[t] = wmma_bf16(load_a(Vblk, 32, 16*t, 0, lane), pb, ot[t]);
  }

  float inv = 1.f / l_run;
  int qcol = lane & 15, half = lane >> 4;
  bf16_t* orow = O + ((size_t)(b * 1024 + q0 + qcol)) * 1024 + h * 64;
#pragma unroll
  for (int t = 0; t < 4; ++t)
#pragma unroll
    for (int r = 0; r < 8; ++r)
      orow[16*t + r + 8*half] = f2bf(ot[t][r] * inv);
}

// ---------------------------------------------------------------------------
// Local attention: 8x16 (b,h), Lq=Lk=8 only (reference slices seq 0..7).
// ---------------------------------------------------------------------------
__global__ __launch_bounds__(256) void attn_local(
    const bf16_t* __restrict__ Q, const bf16_t* __restrict__ Kc,
    const bf16_t* __restrict__ V, bf16_t* __restrict__ O)
{
  int t = blockIdx.x * 256 + threadIdx.x;
  if (t >= 8 * 16 * 8) return;
  int b = t >> 7, rem = t & 127, h = rem >> 3, q = rem & 7;
  const bf16_t* qv = Q + (size_t)(b*8 + q) * 1024 + h * 64;

  float sc[8], mx = -__builtin_inff();
  for (int k = 0; k < 8; ++k) {
    const bf16_t* kv = Kc + (size_t)(b*8 + k) * 1024 + h * 64;
    float s = 0.f;
    for (int d = 0; d < 64; ++d) s += bf2f(qv[d]) * bf2f(kv[d]);
    sc[k] = s; mx = fmaxf(mx, s);
  }
  float l = 0.f;
  for (int k = 0; k < 8; ++k) { sc[k] = __expf(sc[k] - mx); l += sc[k]; }
  float inv = 1.f / l;
  bf16_t* ov = O + (size_t)(b*8 + q) * 1024 + h * 64;
  for (int d = 0; d < 64; ++d) {
    float acc = 0.f;
    for (int k = 0; k < 8; ++k) acc += sc[k] * bf2f(V[(size_t)(b*8 + k)*1024 + h*64 + d]);
    ov[d] = f2bf(acc * inv);
  }
}

// ---------------------------------------------------------------------------
// Host orchestration. Workspace budget ~101 MB.
// ---------------------------------------------------------------------------
extern "C" void kernel_launch(void* const* d_in, const int* in_sizes, int n_in,
                              void* d_out, int out_size, void* d_ws, size_t ws_size,
                              hipStream_t stream) {
  const float* x      = (const float*)d_in[0];
  const float* ln_l_g = (const float*)d_in[1];
  const float* ln_l_b = (const float*)d_in[2];
  const float* ln_g_g = (const float*)d_in[3];
  const float* ln_g_b = (const float*)d_in[4];
  const float* wl_in  = (const float*)d_in[5];
  const float* bl_in  = (const float*)d_in[6];
  const float* wl_out = (const float*)d_in[7];
  const float* bl_out = (const float*)d_in[8];
  const float* wg_in  = (const float*)d_in[9];
  const float* bg_in  = (const float*)d_in[10];
  const float* wg_out = (const float*)d_in[11];
  const float* bg_out = (const float*)d_in[12];
  float* out = (float*)d_out;

  char* ws = (char*)d_ws;
  size_t off = 0;
  auto alloc = [&](size_t bytes) -> char* {
    char* p = ws + off;
    off += (bytes + 255) & ~(size_t)255;
    return p;
  };
  bf16_t* xg   = (bf16_t*)alloc(8192ull * 1024 * 2);
  bf16_t* xloc = (bf16_t*)alloc(64ull   * 1024 * 2);
  bf16_t* wgi  = (bf16_t*)alloc(3072ull * 1024 * 2);
  bf16_t* wgo  = (bf16_t*)alloc(1024ull * 1024 * 2);
  bf16_t* wli  = (bf16_t*)alloc(3072ull * 1024 * 2);
  bf16_t* wlo  = (bf16_t*)alloc(1024ull * 1024 * 2);
  bf16_t* qg   = (bf16_t*)alloc(8192ull * 1024 * 2);
  bf16_t* kg   = (bf16_t*)alloc(8192ull * 1024 * 2);
  bf16_t* vtg  = (bf16_t*)alloc(8192ull * 1024 * 2);
  bf16_t* og   = (bf16_t*)alloc(8192ull * 1024 * 2);
  bf16_t* ql   = (bf16_t*)alloc(64ull   * 1024 * 2);
  bf16_t* kl   = (bf16_t*)alloc(64ull   * 1024 * 2);
  bf16_t* vl   = (bf16_t*)alloc(64ull   * 1024 * 2);
  bf16_t* ol   = (bf16_t*)alloc(64ull   * 1024 * 2);

  // 1. weight conversions to bf16
  cvt_bf16<<<(3072*1024 + 255)/256, 256, 0, stream>>>(wg_in,  wgi, 3072*1024);
  cvt_bf16<<<(1024*1024 + 255)/256, 256, 0, stream>>>(wg_out, wgo, 1024*1024);
  cvt_bf16<<<(3072*1024 + 255)/256, 256, 0, stream>>>(wl_in,  wli, 3072*1024);
  cvt_bf16<<<(1024*1024 + 255)/256, 256, 0, stream>>>(wl_out, wlo, 1024*1024);

  // 2. fused dual layernorm
  ln_kernel<<<8192, 256, 0, stream>>>(x, ln_l_g, ln_l_b, ln_g_g, ln_g_b, xg, xloc);

  // 3. QKV projections (WMMA); dh^-0.5 = 0.125
  gemm_qkv<<<1536, 256, 0, stream>>>(xg,   wgi, bg_in, qg, kg, vtg, 8192, 3072, 1024, 0.125f, 1);
  gemm_qkv<<<12,   256, 0, stream>>>(xloc, wli, bl_in, ql, kl, vl,  64,   3072, 1024, 0.125f, 0);

  // 4. attention
  attn_global<<<1024, 256, 0, stream>>>(qg, kg, vtg, og);
  attn_local <<<4,    256, 0, stream>>>(ql, kl, vl, ol);

  // 5. output projections: global writes d_out, local scatter-adds 64 rows
  gemm_proj<<<512, 256, 0, stream>>>(og, wgo, bg_out, out, 8192, 1024, 1024, 0);
  gemm_proj<<<4,   256, 0, stream>>>(ol, wlo, bl_out, out, 64,   1024, 1024, 1);
}
